// NNUEMCTSModel_29334626631942
// MI455X (gfx1250) — matrix operand order repaired and checked by reading, not compiled, not run on hardware
//
#include <hip/hip_runtime.h>
#include <hip/hip_bf16.h>

// ---------------------------------------------------------------------------
// NNUE forward for MI455X (gfx1250), round 3.
// - layer GEMMs on v_wmma_f32_16x16x32_f16 (f32 accumulate)
// - feature transform as packed f16 (v_pk_add_f16 / v_pk_max_num_f16)
// - two 16-row batch tiles per wave; W1/W2 B-fragments shared across tiles
// - bias staging via global_load_async_to_lds_b32 + s_wait_asynccnt (ASYNCcnt)
// ---------------------------------------------------------------------------

#define BATCH      262144
#define ROWS_PB    256               // rows per block: 8 waves x 32 rows
#define NTILES     (BATCH / ROWS_PB) // 1024
#define KPAD       608               // 578 padded to 19*32
#define W1S        648               // sW1 row stride (halves): 324 dw == 4 mod 64 -> conflict-free
#define FTS        264               // sFT row stride (halves): 132 dw == 4 mod 64 -> conflict-free
#define SHS        40                // sH  row stride (halves): spreads 16 rows over banks

typedef __attribute__((ext_vector_type(16))) _Float16 v16h;
typedef __attribute__((ext_vector_type(8)))  _Float16 v8h;
typedef __attribute__((ext_vector_type(8)))  float    v8f;
typedef __attribute__((ext_vector_type(2)))  float    f2;

struct Smem {
  _Float16 w1[32 * W1S];      // W1 f16, K zero-padded to 608
  _Float16 w2[64 * 32];       // rows 0..59 = W2p, 60 = W2v, 61..63 = 0
  _Float16 ft[120 * FTS];     // ft_w f16, padded row stride
  _Float16 ftb[256];          // ft_b f16
  float    b1[32];            // } contiguous 93-dword bias region,
  float    b2[61];            // } filled by async DMA (b2[60] = b2v)
  _Float16 h[8 * 32 * SHS];   // per-wave h transpose staging (2 tiles x 16 rows)
};

__device__ __forceinline__ v8f wmma16(v16h a, v16h b, v8f c) {
  return __builtin_amdgcn_wmma_f32_16x16x32_f16(false, a, false, b, (short)0, c,
                                                false, false);
}

__device__ __forceinline__ v16h combine(v8h lo, v8h hi) {
  v16h a;
#pragma unroll
  for (int i = 0; i < 8; ++i) { a[i] = lo[i]; a[i + 8] = hi[i]; }
  return a;
}

// relu(ftb + w0 + w1 + w2) fully in packed f16 (12 pk_add + 4 pk_max, no cvts)
__device__ __forceinline__ v8h ft8(const _Float16* p0, const _Float16* p1,
                                   const _Float16* p2, const _Float16* pb, int off) {
  v8h w0 = *(const v8h*)(p0 + off);
  v8h w1 = *(const v8h*)(p1 + off);
  v8h w2 = *(const v8h*)(p2 + off);
  v8h bb = *(const v8h*)(pb + off);
  v8h s = (bb + w0) + (w1 + w2);
  const v8h z = {};
#if __has_builtin(__builtin_elementwise_max)
  return __builtin_elementwise_max(s, z);
#else
  v8h r;
#pragma unroll
  for (int i = 0; i < 8; ++i) r[i] = s[i] > (_Float16)0.0f ? s[i] : (_Float16)0.0f;
  return r;
#endif
}

__device__ __forceinline__ v16h ftfrag(const _Float16* p0, const _Float16* p1,
                                       const _Float16* p2, const _Float16* pb, int off) {
  return combine(ft8(p0, p1, p2, pb, off), ft8(p0, p1, p2, pb, off + 16));
}

__device__ __forceinline__ v8h dense8_full(const float* __restrict__ drow, int dc) {
  v8h r;
#pragma unroll
  for (int i = 0; i < 4; ++i) {
    f2 x = *(const f2*)(drow + dc + 2 * i);   // 8B-aligned, always in-range
    r[2 * i]     = (_Float16)x.x;
    r[2 * i + 1] = (_Float16)x.y;
  }
  return r;
}

__device__ __forceinline__ v8h dense8_part(const float* __restrict__ drow, int dc) {
  v8h r;
#pragma unroll
  for (int i = 0; i < 8; ++i)
    r[i] = (_Float16)(((dc + i) < 66) ? drow[dc + i] : 0.0f);
  return r;
}

// first-3 stm / nstm feature indices, reference _top3_idx semantics
struct Picked { int s0, s1, s2, n0, n1, n2; };
__device__ __forceinline__ Picked pick3(const int* __restrict__ sp, int stmv) {
  Picked p = {0, 0, 0, 0, 0, 0};
  int sc = 0, nc = 0;
#pragma unroll
  for (int j = 0; j < 6; ++j) {
    int f = sp[j];
    bool isp1 = f < 60;
    bool smask = (stmv == 0) ? isp1 : !isp1;  // validity NOT applied to stm (matches ref)
    if (smask) {
      int fc = f < 0 ? 0 : f;                 // clip(idx, 0)
      if (sc == 0) p.s0 = fc; else if (sc == 1) p.s1 = fc; else if (sc == 2) p.s2 = fc;
      ++sc;
    } else if (f >= 0) {                      // nstm requires validity
      if (nc == 0) p.n0 = f; else if (nc == 1) p.n1 = f; else if (nc == 2) p.n2 = f;
      ++nc;
    }
  }
  return p;
}

__global__ __launch_bounds__(256)
void nnue_mcts_fwd(const int*   __restrict__ d_sparse,  // (B,6) int32
                   const float* __restrict__ d_dense,   // (B,66)
                   const int*   __restrict__ d_stm,     // (B,)
                   const float* __restrict__ d_ftw,     // (120,256)
                   const float* __restrict__ d_ftb,     // (256,)
                   const float* __restrict__ d_w1,      // (32,578)
                   const float* __restrict__ d_b1,      // (32,)
                   const float* __restrict__ d_w2v,     // (1,32)
                   const float* __restrict__ d_b2v,     // (1,)
                   const float* __restrict__ d_w2p,     // (60,32)
                   const float* __restrict__ d_b2p,     // (60,)
                   float*       __restrict__ d_out)     // B*60 policy, then B value
{
  __shared__ Smem sm;
  const int tid = threadIdx.x;

  // ---- bias region via async DMA to LDS (CDNA5 ASYNCcnt path) ----
  // sm.b1/sm.b2 form 93 contiguous dwords; per-lane LDS byte address in VDST,
  // 64-bit global address in VADDR (GV mode).
  if (tid < 93) {
    const float* src = (tid < 32) ? (d_b1 + tid)
                     : (tid < 92) ? (d_b2p + (tid - 32))
                                  : d_b2v;
    unsigned int ldsDst = (unsigned int)(uintptr_t)(sm.b1 + tid);
    asm volatile("global_load_async_to_lds_b32 %0, %1, off"
                 :: "v"(ldsDst), "v"((unsigned long long)(uintptr_t)src)
                 : "memory");
  }

  // ---- stage f16 weights to LDS (conversion requires VALU path) ----
  for (int i = tid; i < 32 * KPAD; i += 256) {
    int n = i / KPAD, k = i - n * KPAD;
    sm.w1[n * W1S + k] = (k < 578) ? (_Float16)d_w1[n * 578 + k] : (_Float16)0.0f;
  }
  for (int i = tid; i < 64 * 32; i += 256) {
    int n = i >> 5, k = i & 31;
    float v = (n < 60) ? d_w2p[n * 32 + k] : ((n == 60) ? d_w2v[k] : 0.0f);
    sm.w2[i] = (_Float16)v;
  }
  for (int i = tid; i < 120 * 256; i += 256) {
    int r = i >> 8, c = i & 255;
    sm.ft[r * FTS + c] = (_Float16)d_ftw[i];
  }
  if (tid < 256) sm.ftb[tid] = (_Float16)d_ftb[tid];

  // drain this wave's async loads before publishing via barrier
#if __has_builtin(__builtin_amdgcn_s_wait_asynccnt)
  __builtin_amdgcn_s_wait_asynccnt(0);
#else
  asm volatile("s_wait_asynccnt 0" ::: "memory");
#endif
  __syncthreads();

  const int lane = tid & 31;
  const int wave = tid >> 5;
  const int half = lane >> 4;
  const int l15  = lane & 15;
  _Float16* sHw = sm.h + wave * (32 * SHS);

  // tile-invariant LDS bases (chunk offsets fold into ds immediates)
  const _Float16* w1b0 = sm.w1 + l15 * W1S + 8 * half;
  const _Float16* w1b1 = sm.w1 + (16 + l15) * W1S + 8 * half;
  const _Float16* ftbb = sm.ftb + 8 * half;

  for (int tile = blockIdx.x; tile < NTILES; tile += (int)gridDim.x) {
    const int rowBase = tile * ROWS_PB + wave * 32;
    const int rowA = rowBase + l15;        // tile part A
    const int rowB = rowA + 16;            // tile part B

    { // prefetch next tile's per-row streams (global_prefetch_b8)
      int pt = tile + (int)gridDim.x;
      if (pt < NTILES) {
        int pr = pt * ROWS_PB + wave * 32 + l15;
        __builtin_prefetch(d_sparse + pr * 6, 0, 3);
        __builtin_prefetch(d_dense + pr * 66, 0, 3);
        __builtin_prefetch(d_sparse + (pr + 16) * 6, 0, 3);
        __builtin_prefetch(d_dense + (pr + 16) * 66, 0, 3);
      }
    }

    const Picked pA = pick3(d_sparse + rowA * 6, d_stm[rowA]);
    const Picked pB = pick3(d_sparse + rowB * 6, d_stm[rowB]);

    // per-row ft_w row bases (+8*half column phase)
    const _Float16* As0 = sm.ft + pA.s0 * FTS + 8 * half;
    const _Float16* As1 = sm.ft + pA.s1 * FTS + 8 * half;
    const _Float16* As2 = sm.ft + pA.s2 * FTS + 8 * half;
    const _Float16* An0 = sm.ft + pA.n0 * FTS + 8 * half;
    const _Float16* An1 = sm.ft + pA.n1 * FTS + 8 * half;
    const _Float16* An2 = sm.ft + pA.n2 * FTS + 8 * half;
    const _Float16* Bs0 = sm.ft + pB.s0 * FTS + 8 * half;
    const _Float16* Bs1 = sm.ft + pB.s1 * FTS + 8 * half;
    const _Float16* Bs2 = sm.ft + pB.s2 * FTS + 8 * half;
    const _Float16* Bn0 = sm.ft + pB.n0 * FTS + 8 * half;
    const _Float16* Bn1 = sm.ft + pB.n1 * FTS + 8 * half;
    const _Float16* Bn2 = sm.ft + pB.n2 * FTS + 8 * half;

    const float* drowA = d_dense + rowA * 66;
    const float* drowB = d_dense + rowB * 66;

    v8f c0a = {}, c1a = {}, c0b = {}, c1b = {};

    // ---- layer 1, K-chunks 0..7: stm feature-transform ----
#pragma unroll
    for (int c = 0; c < 8; ++c) {
      const v16h bf0 = combine(*(const v8h*)(w1b0 + 32 * c), *(const v8h*)(w1b0 + 32 * c + 16));
      const v16h bf1 = combine(*(const v8h*)(w1b1 + 32 * c), *(const v8h*)(w1b1 + 32 * c + 16));
      const v16h aA = ftfrag(As0, As1, As2, ftbb, 32 * c);
      const v16h aB = ftfrag(Bs0, Bs1, Bs2, ftbb, 32 * c);
      c0a = wmma16(aA, bf0, c0a);  c1a = wmma16(aA, bf1, c1a);
      c0b = wmma16(aB, bf0, c0b);  c1b = wmma16(aB, bf1, c1b);
    }
    // ---- K-chunks 8..15: nstm feature-transform ----
#pragma unroll
    for (int c = 0; c < 8; ++c) {
      const v16h bf0 = combine(*(const v8h*)(w1b0 + 32 * (8 + c)), *(const v8h*)(w1b0 + 32 * (8 + c) + 16));
      const v16h bf1 = combine(*(const v8h*)(w1b1 + 32 * (8 + c)), *(const v8h*)(w1b1 + 32 * (8 + c) + 16));
      const v16h aA = ftfrag(An0, An1, An2, ftbb, 32 * c);
      const v16h aB = ftfrag(Bn0, Bn1, Bn2, ftbb, 32 * c);
      c0a = wmma16(aA, bf0, c0a);  c1a = wmma16(aA, bf1, c1a);
      c0b = wmma16(aB, bf0, c0b);  c1b = wmma16(aB, bf1, c1b);
    }
    // ---- K-chunks 16..17: dense features (fully in-range) ----
#pragma unroll
    for (int g = 16; g < 18; ++g) {
      const int dlo = 8 * (4 * g + half) - 512;
      const v16h bf0 = combine(*(const v8h*)(w1b0 + 32 * g), *(const v8h*)(w1b0 + 32 * g + 16));
      const v16h bf1 = combine(*(const v8h*)(w1b1 + 32 * g), *(const v8h*)(w1b1 + 32 * g + 16));
      const v16h aA = combine(dense8_full(drowA, dlo), dense8_full(drowA, dlo + 16));
      const v16h aB = combine(dense8_full(drowB, dlo), dense8_full(drowB, dlo + 16));
      c0a = wmma16(aA, bf0, c0a);  c1a = wmma16(aA, bf1, c1a);
      c0b = wmma16(aB, bf0, c0b);  c1b = wmma16(aB, bf1, c1b);
    }
    { // ---- K-chunk 18: dense tail + zero pad ----
      const int dlo = 8 * (4 * 18 + half) - 512;
      const v16h bf0 = combine(*(const v8h*)(w1b0 + 32 * 18), *(const v8h*)(w1b0 + 32 * 18 + 16));
      const v16h bf1 = combine(*(const v8h*)(w1b1 + 32 * 18), *(const v8h*)(w1b1 + 32 * 18 + 16));
      const v16h aA = combine(dense8_part(drowA, dlo), dense8_part(drowA, dlo + 16));
      const v16h aB = combine(dense8_part(drowB, dlo), dense8_part(drowB, dlo + 16));
      c0a = wmma16(aA, bf0, c0a);  c1a = wmma16(aA, bf1, c1a);
      c0b = wmma16(aB, bf0, c0b);  c1b = wmma16(aB, bf1, c1b);
    }

    // ---- epilogue: h = relu(c + b1) in f32 -> f16, transpose via LDS ----
#pragma unroll
    for (int part = 0; part < 2; ++part) {
      _Float16* sHp = sHw + part * (16 * SHS);
#pragma unroll
      for (int t = 0; t < 2; ++t) {
        const v8f cc = part ? (t ? c1b : c0b) : (t ? c1a : c0a);
        const int n = 16 * t + l15;
        const float bb = sm.b1[n];
#pragma unroll
        for (int v = 0; v < 8; ++v) {
          float hv = cc[v] + bb;
          hv = hv > 0.0f ? hv : 0.0f;
          sHp[(v + 8 * half) * SHS + n] = (_Float16)hv;
        }
      }
    }

    // ---- layer 2: h(16x32) @ W2^T (61 outputs padded to 64), K = 32 ----
    // B fragments identical for both sub-tiles: load once, feed 8 WMMAs.
    v16h b2f[4];
#pragma unroll
    for (int t = 0; t < 4; ++t)
      b2f[t] = *(const v16h*)(sm.w2 + (16 * t + l15) * 32 + 16 * half);

    const v8f zacc = {};
#pragma unroll
    for (int part = 0; part < 2; ++part) {
      const _Float16* sHp = sHw + part * (16 * SHS);
      const v8h p = *(const v8h*)(sHp + l15 * SHS + 16 * half);
      const v8h q = *(const v8h*)(sHp + l15 * SHS + 16 * half + 8);
      const v16h a2 = combine(p, q);
#pragma unroll
      for (int t = 0; t < 4; ++t) {
        v8f o = wmma16(a2, b2f[t], zacc);
        const int n = 16 * t + l15;
#pragma unroll
        for (int v = 0; v < 8; ++v) {
          const int r = rowBase + 16 * part + v + 8 * half;
          float val = o[v];
          if (n < 60) {
            d_out[r * 60 + n] = val + sm.b2[n];              // policy logits
          } else if (n == 60) {
            d_out[BATCH * 60 + r] = tanhf(val + sm.b2[60]);  // value head
          }
        }
      }
    }
  }
}

extern "C" void kernel_launch(void* const* d_in, const int* in_sizes, int n_in,
                              void* d_out, int out_size, void* d_ws, size_t ws_size,
                              hipStream_t stream) {
  (void)in_sizes; (void)n_in; (void)d_ws; (void)ws_size; (void)out_size;
  const int*   sparse = (const int*)  d_in[0];
  const float* dense  = (const float*)d_in[1];
  const int*   stm    = (const int*)  d_in[2];
  const float* ftw    = (const float*)d_in[3];
  const float* ftb    = (const float*)d_in[4];
  const float* W1     = (const float*)d_in[5];
  const float* b1     = (const float*)d_in[6];
  const float* W2v    = (const float*)d_in[7];
  const float* b2v    = (const float*)d_in[8];
  const float* W2p    = (const float*)d_in[9];
  const float* b2p    = (const float*)d_in[10];
  float* out = (float*)d_out;

  dim3 grid(512), block(256);  // 1024 tiles, grid-stride x2 amortizes LDS staging
  nnue_mcts_fwd<<<grid, block, 0, stream>>>(sparse, dense, stm, ftw, ftb,
                                            W1, b1, W2v, b2v, W2p, b2p, out);
}
